// SparseMoE_77721728189137
// MI455X (gfx1250) — matrix operand order repaired and checked by reading, not compiled, not run on hardware
//
#include <hip/hip_runtime.h>

typedef __bf16 bf16_t;
typedef __attribute__((ext_vector_type(16))) __bf16 v16bf;
typedef __attribute__((ext_vector_type(8)))  __bf16 v8bf;
typedef __attribute__((ext_vector_type(2)))  __bf16 v2bf;
typedef __attribute__((ext_vector_type(8)))  float  v8f;

typedef __attribute__((ext_vector_type(4))) unsigned int tdm_v4u;
typedef __attribute__((ext_vector_type(8))) int          tdm_v8i;
typedef __attribute__((ext_vector_type(4))) int          tdm_v4i;

#define T_TOKENS 4096
#define DM 1024
#define DH 4096
#define NEXP 8
#define NSLOTS (2 * T_TOKENS)   // every token goes to exactly 2 experts

// ---- workspace layout (bytes) ----
// [0,128): control: counts[8] (int), offs[8] (int), cursor[8] (int), psum[8] (float)
#define WSO_SLOT_TOK 128u
#define WSO_SLOT_W   (WSO_SLOT_TOK + 2u*T_TOKENS*4u)     // 32896
#define WSO_TOPI     (WSO_SLOT_W   + 2u*T_TOKENS*4u)     // 65664
#define WSO_TOPW     (WSO_TOPI     + 2u*T_TOKENS*4u)     // 98432
#define WSO_X16      (WSO_TOPW     + 2u*T_TOKENS*4u)     // 131200 (16B aligned)
#define WSO_H        (WSO_X16 + (size_t)T_TOKENS*DM*2u)  // 8519808
#define WSO_XG       (WSO_H   + (size_t)NSLOTS*DH*2u)    // 75628672
// Xg: bf16[NSLOTS][DM] = 16 MiB; total ws use ~92.4 MB

// ---- Tensor Data Mover: 2D bf16 tile (tile_d0 x tile_d1) from global -> LDS ----
// D# per CDNA5 ISA ch.8. Rows >= tensor_d1 are zero-filled (OOB reads return 0).
static __device__ __forceinline__ void tdm_load_2d_bf16(
    unsigned lds_addr, const void* gaddr,
    unsigned tensor_d0, unsigned tensor_d1,
    unsigned tile_d0, unsigned tile_d1, unsigned stride_elems)
{
    const unsigned long long ga = (unsigned long long)(size_t)gaddr;
    tdm_v4u g0;
    g0[0] = 0x1u;                                          // count=1, user mode
    g0[1] = lds_addr;                                      // LDS byte address
    g0[2] = (unsigned)(ga & 0xFFFFFFFFu);                  // global_addr[31:0]
    g0[3] = (unsigned)((ga >> 32) & 0x01FFFFFFu)           // global_addr[56:32]
          | 0x80000000u;                                   // type=2 ("image")
    tdm_v8i g1;
    g1[0] = (int)(1u << 16);                               // wg_mask=0, data_size=1 (2B)
    g1[1] = (int)((tensor_d0 & 0xFFFFu) << 16);            // tensor_dim0[15:0]
    g1[2] = (int)(((tensor_d0 >> 16) & 0xFFFFu)
          |       ((tensor_d1 & 0xFFFFu) << 16));          // td0[31:16] | td1[15:0]
    g1[3] = (int)(((tensor_d1 >> 16) & 0xFFFFu)
          |       ((tile_d0 & 0xFFFFu) << 16));            // td1[31:16] | tile_dim0
    g1[4] = (int)(tile_d1 & 0xFFFFu);                      // tile_dim1 (tile_dim2=0)
    g1[5] = (int)stride_elems;                             // tensor_dim0_stride[31:0]
    g1[6] = 0;                                             // stride[47:32] | dim1_stride lo
    g1[7] = 0;
    const tdm_v4i gz4 = {0, 0, 0, 0};
    const tdm_v8i gz8 = {0, 0, 0, 0, 0, 0, 0, 0};
    __builtin_amdgcn_tensor_load_to_lds(g0, g1, gz4, gz4, gz8, 0);
}

// ---- shared WMMA mainloop: C[128 x 64] += A[128 x KDIM] * W[KDIM x NDIM] tile ----
// A rows contiguous bf16 (TDM double-buffered); W f32 row-major, converted to
// packed-bf16 [N][K] tiles in LDS. acc layout: 8 waves as 4(M) x 2(N), each
// wave 32x32 = 2x2 WMMA 16x16 tiles.
template<int KDIM, int NDIM>
static __device__ __forceinline__ void moe_mainloop(
    const bf16_t* __restrict__ Arows, const float* __restrict__ W,
    int n0, unsigned rows_valid, bf16_t* sA, bf16_t* sB,
    int tid, v8f (&acc)[2][2])
{
    const int wave = tid >> 5, lane = tid & 31;
    const int wm = wave >> 1, wn = wave & 1;
    const int lr = lane & 15, lh = lane >> 4;
    const unsigned sA_lds = (unsigned)(size_t)(void*)sA;

    if (wave == 0)   // preload K-tile 0 into buffer 0
        tdm_load_2d_bf16(sA_lds, Arows, 32u, rows_valid, 32u, 128u, (unsigned)KDIM);

    for (int k0 = 0; k0 < KDIM; k0 += 32) {
        const int cur = (k0 >> 5) & 1;
        if (wave == 0 && (k0 + 32) < KDIM)   // prefetch next K-tile into other buffer
            tdm_load_2d_bf16(sA_lds + (unsigned)((cur ^ 1) * 128 * 32 * 2),
                             Arows + (k0 + 32), 32u, rows_valid, 32u, 128u,
                             (unsigned)KDIM);
        // B: 32(K) x 64(N) f32 tile -> packed bf16 pairs, layout [N][K].
        // Thread owns a K-pair x N-quad: one dword store per (n, k-pair).
        {
            const int nq = (tid & 15) * 4;     // n: 0..60 step 4
            const int kp = (tid >> 4) * 2;     // k: 0..30 step 2
            const float* src0 = W + (size_t)(k0 + kp) * NDIM + n0 + nq;
            const float4 f0 = *(const float4*)src0;
            const float4 f1 = *(const float4*)(src0 + NDIM);
            v2bf p0; p0[0] = (bf16_t)f0.x; p0[1] = (bf16_t)f1.x;
            v2bf p1; p1[0] = (bf16_t)f0.y; p1[1] = (bf16_t)f1.y;
            v2bf p2; p2[0] = (bf16_t)f0.z; p2[1] = (bf16_t)f1.z;
            v2bf p3; p3[0] = (bf16_t)f0.w; p3[1] = (bf16_t)f1.w;
            *(v2bf*)&sB[(nq + 0) * 32 + kp] = p0;
            *(v2bf*)&sB[(nq + 1) * 32 + kp] = p1;
            *(v2bf*)&sB[(nq + 2) * 32 + kp] = p2;
            *(v2bf*)&sB[(nq + 3) * 32 + kp] = p3;
        }
        if (k0 + 32 < KDIM)
            __builtin_prefetch(W + (size_t)(k0 + 32) * NDIM + n0, 0, 1);
        if (wave == 0) {
            if ((k0 + 32) < KDIM) __builtin_amdgcn_s_wait_tensorcnt(1); // cur done,
            else                  __builtin_amdgcn_s_wait_tensorcnt(0); // next in flight
        }
        __syncthreads();

        const bf16_t* sAc = sA + cur * (128 * 32);
        #pragma unroll
        for (int tm = 0; tm < 2; tm++) {
            const int row = wm * 32 + tm * 16 + lr;
            const v8bf lo = *(const v8bf*)&sAc[row * 32 + lh * 8];
            const v8bf hi = *(const v8bf*)&sAc[row * 32 + lh * 8 + 16];
            const v16bf a = __builtin_shufflevector(lo, hi,
                0,1,2,3,4,5,6,7,8,9,10,11,12,13,14,15);
            #pragma unroll
            for (int tn = 0; tn < 2; tn++) {
                const int col = wn * 32 + tn * 16 + lr;
                const v8bf blo = *(const v8bf*)&sB[col * 32 + lh * 8];
                const v8bf bhi = *(const v8bf*)&sB[col * 32 + lh * 8 + 16];
                const v16bf b = __builtin_shufflevector(blo, bhi,
                    0,1,2,3,4,5,6,7,8,9,10,11,12,13,14,15);
                acc[tm][tn] = __builtin_amdgcn_wmma_f32_16x16x32_bf16(
                    false, a, false, b, (short)0, acc[tm][tn], false, false);
            }
        }
        __syncthreads();
    }
}

__global__ void k_zero(float* __restrict__ out, int out_n, int* __restrict__ ctrl) {
    int i = blockIdx.x * blockDim.x + threadIdx.x;
    for (int idx = i; idx < out_n; idx += gridDim.x * blockDim.x) out[idx] = 0.0f;
    if (i < 32) ctrl[i] = 0;
}

// One wave per token: router logits, softmax, top-2, counts, prob sums, x -> bf16.
__global__ __launch_bounds__(256) void k_router(
    const float* __restrict__ x, const float* __restrict__ rw,
    const float* __restrict__ rb, bf16_t* __restrict__ x16,
    int* __restrict__ counts, float* __restrict__ psum,
    int* __restrict__ topi, float* __restrict__ topw)
{
    const int wave = threadIdx.x >> 5;
    const int lane = threadIdx.x & 31;
    const int t = blockIdx.x * 8 + wave;
    const float* xr = x + (size_t)t * DM;

    float acc[NEXP];
    #pragma unroll
    for (int e = 0; e < NEXP; e++) acc[e] = 0.0f;

    #pragma unroll 4
    for (int i = 0; i < DM / 32; i++) {
        const int d = lane + 32 * i;
        const float xv = xr[d];
        const float* w = rw + (size_t)d * NEXP;
        #pragma unroll
        for (int e = 0; e < NEXP; e++) acc[e] += xv * w[e];
        x16[(size_t)t * DM + d] = (bf16_t)xv;
    }
    #pragma unroll
    for (int off = 16; off > 0; off >>= 1) {
        #pragma unroll
        for (int e = 0; e < NEXP; e++) acc[e] += __shfl_xor(acc[e], off, 32);
    }
    float probs[NEXP];
    float mx = -1e30f;
    #pragma unroll
    for (int e = 0; e < NEXP; e++) { acc[e] += rb[e]; mx = fmaxf(mx, acc[e]); }
    float s = 0.0f;
    #pragma unroll
    for (int e = 0; e < NEXP; e++) { probs[e] = __expf(acc[e] - mx); s += probs[e]; }
    const float inv = 1.0f / s;
    #pragma unroll
    for (int e = 0; e < NEXP; e++) probs[e] *= inv;

    int i0 = 0;
    #pragma unroll
    for (int e = 1; e < NEXP; e++) if (probs[e] > probs[i0]) i0 = e;
    int i1 = (i0 == 0) ? 1 : 0;
    #pragma unroll
    for (int e = 0; e < NEXP; e++) if (e != i0 && probs[e] > probs[i1]) i1 = e;
    const float v0 = probs[i0], v1 = probs[i1];
    const float inv2 = 1.0f / (v0 + v1);

    if (lane == 0) {
        topi[t * 2 + 0] = i0; topi[t * 2 + 1] = i1;
        topw[t * 2 + 0] = v0 * inv2; topw[t * 2 + 1] = v1 * inv2;
        atomicAdd(&counts[i0], 1);
        atomicAdd(&counts[i1], 1);
    }
    if (lane < NEXP) atomicAdd(&psum[lane], probs[lane]);
}

__global__ void k_scan(const int* __restrict__ counts, int* __restrict__ offs) {
    if (threadIdx.x == 0) {
        int o = 0;
        for (int e = 0; e < NEXP; e++) { offs[e] = o; o += counts[e]; }
    }
}

__global__ void k_assign(const int* __restrict__ topi, const float* __restrict__ topw,
                         const int* __restrict__ offs, int* __restrict__ cursor,
                         int* __restrict__ stok, float* __restrict__ swt)
{
    const int t = blockIdx.x * blockDim.x + threadIdx.x;
    if (t >= T_TOKENS) return;
    for (int k = 0; k < 2; k++) {
        const int e = topi[t * 2 + k];
        const int pos = atomicAdd(&cursor[e], 1);
        const int slot = offs[e] + pos;
        stok[slot] = t;
        swt[slot] = topw[t * 2 + k];
    }
}

// Compact x16 rows into slot order so GEMM1's A tiles are contiguous (TDM-able).
// One wave per slot row (2 KB copy).
__global__ __launch_bounds__(256) void k_gather(
    const bf16_t* __restrict__ x16, const int* __restrict__ stok,
    bf16_t* __restrict__ Xg)
{
    const int wave = threadIdx.x >> 5, lane = threadIdx.x & 31;
    const int s = blockIdx.x * 8 + wave;
    const int tok = stok[s];
    const uint4* src = (const uint4*)(x16 + (size_t)tok * DM);
    uint4* dst = (uint4*)(Xg + (size_t)s * DM);
    #pragma unroll
    for (int i = 0; i < 4; i++) dst[lane + 32 * i] = src[lane + 32 * i];
}

// GEMM1: H[slot, n] = gelu(Xg[slot] @ w1[e] + b1[e])
__global__ __launch_bounds__(256) void k_gemm1(
    const bf16_t* __restrict__ Xg, const float* __restrict__ w1,
    const float* __restrict__ b1, const int* __restrict__ counts,
    const int* __restrict__ offs, bf16_t* __restrict__ H)
{
    const int e = blockIdx.z;
    const int cnt = counts[e];
    const int m0 = blockIdx.y * 128;
    if (m0 >= cnt) return;
    const int n0 = blockIdx.x * 64;
    const int base = offs[e];

    __shared__ __align__(16) bf16_t sA[2 * 128 * 32];   // TDM double buffer
    __shared__ __align__(16) bf16_t sB[64 * 32];

    const int tid = threadIdx.x;
    const int wave = tid >> 5, lane = tid & 31;
    const int wm = wave >> 1, wn = wave & 1;
    const int lr = lane & 15, lh = lane >> 4;

    v8f acc[2][2] = {};
    moe_mainloop<DM, DH>(Xg + (size_t)(base + m0) * DM,
                         w1 + (size_t)e * DM * DH, n0,
                         (unsigned)(cnt - m0), sA, sB, tid, acc);

    const float kInvSqrt2 = 0.70710678118654752f;
    #pragma unroll
    for (int tm = 0; tm < 2; tm++) {
        #pragma unroll
        for (int tn = 0; tn < 2; tn++) {
            const int colT = wn * 32 + tn * 16 + lr;
            const float bias = b1[(size_t)e * DH + n0 + colT];
            #pragma unroll
            for (int r = 0; r < 8; r++) {
                const int rowT = wm * 32 + tm * 16 + lh * 8 + r;
                const int grow = m0 + rowT;
                if (grow < cnt) {
                    const float v = acc[tm][tn][r] + bias;
                    const float g = 0.5f * v * (1.0f + erff(v * kInvSqrt2));
                    H[(size_t)(base + grow) * DH + n0 + colT] = (bf16_t)g;
                }
            }
        }
    }
}

// GEMM2: out[tok] += w_slot * (H[slot] @ w2[e] + b2[e])
__global__ __launch_bounds__(256) void k_gemm2(
    const bf16_t* __restrict__ H, const float* __restrict__ w2,
    const float* __restrict__ b2, const int* __restrict__ counts,
    const int* __restrict__ offs, const int* __restrict__ stok,
    const float* __restrict__ swt, float* __restrict__ out)
{
    const int e = blockIdx.z;
    const int cnt = counts[e];
    const int m0 = blockIdx.y * 128;
    if (m0 >= cnt) return;
    const int n0 = blockIdx.x * 64;
    const int base = offs[e];

    __shared__ __align__(16) bf16_t sA[2 * 128 * 32];   // TDM double buffer
    __shared__ __align__(16) bf16_t sB[64 * 32];
    __shared__ int sTok[128];
    __shared__ float sW[128];

    const int tid = threadIdx.x;
    if (tid < 128) {
        const int grow = m0 + tid;
        const int slot = (grow < cnt) ? (base + grow) : base;
        sTok[tid] = stok[slot];
        sW[tid] = (grow < cnt) ? swt[slot] : 0.0f;
    }
    __syncthreads();

    const int wave = tid >> 5, lane = tid & 31;
    const int wm = wave >> 1, wn = wave & 1;
    const int lr = lane & 15, lh = lane >> 4;

    v8f acc[2][2] = {};
    moe_mainloop<DH, DM>(H + (size_t)(base + m0) * DH,
                         w2 + (size_t)e * DH * DM, n0,
                         (unsigned)(cnt - m0), sA, sB, tid, acc);

    #pragma unroll
    for (int tm = 0; tm < 2; tm++) {
        #pragma unroll
        for (int tn = 0; tn < 2; tn++) {
            const int colT = wn * 32 + tn * 16 + lr;
            const float bias = b2[(size_t)e * DM + n0 + colT];
            #pragma unroll
            for (int r = 0; r < 8; r++) {
                const int rowT = wm * 32 + tm * 16 + lh * 8 + r;
                const int grow = m0 + rowT;
                if (grow < cnt) {
                    const float val = (acc[tm][tn][r] + bias) * sW[rowT];
                    atomicAdd(&out[(size_t)sTok[rowT] * DM + n0 + colT], val);
                }
            }
        }
    }
}

__global__ void k_aux(const float* __restrict__ psum, float* __restrict__ out_aux) {
    if (threadIdx.x == 0 && blockIdx.x == 0) {
        float mp[NEXP], m = 0.0f;
        for (int e = 0; e < NEXP; e++) { mp[e] = psum[e] / (float)T_TOKENS; m += mp[e]; }
        m /= (float)NEXP;
        float v = 0.0f;
        for (int e = 0; e < NEXP; e++) { const float d = mp[e] - m; v += d * d; }
        out_aux[0] = v / (float)(NEXP - 1);   // ddof=1
    }
}

extern "C" void kernel_launch(void* const* d_in, const int* in_sizes, int n_in,
                              void* d_out, int out_size, void* d_ws, size_t ws_size,
                              hipStream_t stream) {
    const float* x  = (const float*)d_in[0];
    const float* rw = (const float*)d_in[1];
    const float* rb = (const float*)d_in[2];
    const float* w1 = (const float*)d_in[3];
    const float* b1 = (const float*)d_in[4];
    const float* w2 = (const float*)d_in[5];
    const float* b2 = (const float*)d_in[6];
    float* out = (float*)d_out;

    char* ws = (char*)d_ws;
    int*   counts = (int*)(ws + 0);
    int*   offs   = (int*)(ws + 32);
    int*   cursor = (int*)(ws + 64);
    float* psum   = (float*)(ws + 96);
    int*   stok   = (int*)(ws + WSO_SLOT_TOK);
    float* swt    = (float*)(ws + WSO_SLOT_W);
    int*   topi   = (int*)(ws + WSO_TOPI);
    float* topw   = (float*)(ws + WSO_TOPW);
    bf16_t* x16   = (bf16_t*)(ws + WSO_X16);
    bf16_t* Hbuf  = (bf16_t*)(ws + WSO_H);
    bf16_t* Xg    = (bf16_t*)(ws + WSO_XG);

    k_zero<<<1024, 256, 0, stream>>>(out, out_size, (int*)ws);
    k_router<<<T_TOKENS / 8, 256, 0, stream>>>(x, rw, rb, x16, counts, psum, topi, topw);
    k_scan<<<1, 32, 0, stream>>>(counts, offs);
    k_assign<<<T_TOKENS / 256, 256, 0, stream>>>(topi, topw, offs, cursor, stok, swt);
    k_gather<<<NSLOTS / 8, 256, 0, stream>>>(x16, stok, Xg);

    dim3 g1(DH / 64, T_TOKENS / 128, NEXP);   // 64 x 32 x 8
    k_gemm1<<<g1, 256, 0, stream>>>(Xg, w1, b1, counts, offs, Hbuf);

    dim3 g2(DM / 64, T_TOKENS / 128, NEXP);   // 16 x 32 x 8
    k_gemm2<<<g2, 256, 0, stream>>>(Hbuf, w2, b2, counts, offs, stok, swt, out);

    k_aux<<<1, 32, 0, stream>>>(psum, out + (size_t)T_TOKENS * DM);
}
